// EncoderLayer_86045374808931
// MI455X (gfx1250) — matrix operand order repaired and checked by reading, not compiled, not run on hardware
//
#include <hip/hip_runtime.h>
#include <hip/hip_bf16.h>
#include <stdint.h>

// ---------------------------------------------------------------------------
// Problem constants (match reference)
// ---------------------------------------------------------------------------
#define BATCH   4
#define SEQ     2048
#define DMODEL  512
#define NHEADS  8
#define DHEAD   64
#define DFF     2048
#define MTOT    (BATCH * SEQ)          // 8192 rows

typedef __bf16 bf16_t;
typedef __attribute__((ext_vector_type(16))) __bf16 bf16x16;
typedef __attribute__((ext_vector_type(8)))  float  f32x8;

// CDNA5 async global->LDS path (ASYNCcnt-tracked), with sync fallback.
#if __has_builtin(__builtin_amdgcn_global_load_async_to_lds_b128) && \
    __has_builtin(__builtin_amdgcn_s_wait_asynccnt)
#define HAVE_ASYNC_LDS 1
typedef int v4i __attribute__((vector_size(16)));
typedef __attribute__((address_space(1))) v4i as1_v4i;   // global
typedef __attribute__((address_space(3))) v4i as3_v4i;   // LDS
#else
#define HAVE_ASYNC_LDS 0
#endif

// ---------------------------------------------------------------------------
// WMMA helpers (wave32, 16x16x32 bf16 -> f32)
// ---------------------------------------------------------------------------
__device__ __forceinline__ f32x8 wmma_bf16(bf16x16 a, bf16x16 b, f32x8 c) {
  return __builtin_amdgcn_wmma_f32_16x16x32_bf16(
      /*neg_a=*/false, a, /*neg_b=*/false, b,
      /*c_mod=*/(short)0, c, /*reuse_a=*/false, /*reuse_b=*/false);
}

// A fragment: 16x32 (MxK). lane%16 = M; element e holds
//   K = (lane>=16 ? 16 : 0) + e
__device__ __forceinline__ bf16x16 load_frag_a(const bf16_t* lds, int stride,
                                               int mbase, int lane) {
  bf16x16 v;
  const bf16_t* p = lds + (size_t)(mbase + (lane & 15)) * stride;
  const int kh = (lane >> 4) << 4;   // 0 or 16
#pragma unroll
  for (int e = 0; e < 16; ++e)
    v[e] = p[kh + e];
  return v;
}

// B fragment: 32x16 (KxN), tile stored [N][K] in LDS.
//   N = lane&15, element e holds K = (lane>=16 ? 16 : 0) + e
__device__ __forceinline__ bf16x16 load_frag_b(const bf16_t* lds, int stride,
                                               int nbase, int lane) {
  bf16x16 v;
  const bf16_t* p = lds + (size_t)(nbase + (lane & 15)) * stride + ((lane >> 4) << 4);
#pragma unroll
  for (int e = 0; e < 16; ++e) v[e] = p[e];
  return v;
}

// ---------------------------------------------------------------------------
// f32 -> bf16 conversion
// ---------------------------------------------------------------------------
__global__ void cvt_f32_bf16_kernel(const float* __restrict__ src,
                                    bf16_t* __restrict__ dst, int n) {
  int i = blockIdx.x * 256 + threadIdx.x;
  if (i < n) dst[i] = (bf16_t)src[i];
}

// ---------------------------------------------------------------------------
// Generic GEMM: C[M,N] = A[M,K](bf16, row-major) x B[K,N](bf16, row-major)
// Block tile 128x128, BK=32, double-buffered LDS.
// 256 threads = 8 waves in 4(M) x 2(N); each wave computes 32x64 via
// 2x4 WMMA fragments (8 wmma per K-step per wave).
// A tile staged via async global->LDS B128 when available.
// ---------------------------------------------------------------------------
#define GBM 128
#define GBN 128
#define GBK 32
#define GLDA 40                 // 80B row stride: keeps B128 chunks 16B-aligned
#define GLDB 34
#define NBUF_A (GBM * GLDA)
#define NBUF_B (GBN * GLDB)

template <bool RELU>
__global__ __launch_bounds__(256)
void gemm_bf16_kernel(const bf16_t* __restrict__ A, const bf16_t* __restrict__ B,
                      float* __restrict__ Cf, bf16_t* __restrict__ Cb,
                      int M, int N, int K) {
  __shared__ bf16_t As[2 * NBUF_A];
  __shared__ bf16_t Bs[2 * NBUF_B];

  const int tid  = threadIdx.x;
  const int lane = tid & 31;
  const int wid  = tid >> 5;
  const int wm   = wid >> 1;          // 0..3
  const int wn   = wid & 1;           // 0..1
  const int m0   = blockIdx.y * GBM;
  const int n0   = blockIdx.x * GBN;

  auto load_tile = [&](int k0, int buf) {
    bf16_t* as = As + buf * NBUF_A;
    bf16_t* bs = Bs + buf * NBUF_B;
    // A tile: 128x32 bf16 = 512 B128 chunks (4 per row), 2 chunks/thread.
#pragma unroll
    for (int t = 0; t < 2; ++t) {
      int c   = tid * 2 + t;
      int r   = c >> 2;
      int sub = (c & 3) << 3;         // element offset within row
      const bf16_t* g = &A[(size_t)(m0 + r) * K + k0 + sub];
      bf16_t*       l = &as[r * GLDA + sub];
#if HAVE_ASYNC_LDS
      __builtin_amdgcn_global_load_async_to_lds_b128(
          (as1_v4i*)g, (as3_v4i*)l, /*offset=*/0, /*cpol=*/0);
#else
      *(uint4*)l = *(const uint4*)g;
#endif
    }
    // B tile transposed into [N][K]: Bs[n][k] = B[(k0+k)*N + n0+n]
#pragma unroll
    for (int i = 0; i < 16; ++i) {
      int q = tid + i * 256;
      int n = q & 127;
      int k = q >> 7;
      bs[n * GLDB + k] = B[(size_t)(k0 + k) * N + n0 + n];
    }
  };

  f32x8 acc[2][4] = {};
  const int NT = K / GBK;

  load_tile(0, 0);
#if HAVE_ASYNC_LDS
  __builtin_amdgcn_s_wait_asynccnt(0);
#endif
  __syncthreads();

  for (int kt = 0; kt < NT; ++kt) {
    const bf16_t* as = As + (kt & 1) * NBUF_A;
    const bf16_t* bs = Bs + (kt & 1) * NBUF_B;

    if (kt + 1 < NT) load_tile((kt + 1) * GBK, (kt + 1) & 1);

    bf16x16 a[2], b[4];
#pragma unroll
    for (int i = 0; i < 2; ++i) a[i] = load_frag_a(as, GLDA, wm * 32 + i * 16, lane);
#pragma unroll
    for (int j = 0; j < 4; ++j) b[j] = load_frag_b(bs, GLDB, wn * 64 + j * 16, lane);
#pragma unroll
    for (int i = 0; i < 2; ++i)
#pragma unroll
      for (int j = 0; j < 4; ++j)
        acc[i][j] = wmma_bf16(a[i], b[j], acc[i][j]);

    if (kt + 1 < NT) {
#if HAVE_ASYNC_LDS
      __builtin_amdgcn_s_wait_asynccnt(0);
#endif
      __syncthreads();
    }
  }

  const int rhi = (lane >> 4) << 3;   // 0 or 8
  const int cl  = lane & 15;
#pragma unroll
  for (int i = 0; i < 2; ++i)
#pragma unroll
    for (int j = 0; j < 4; ++j)
#pragma unroll
      for (int r = 0; r < 8; ++r) {
        int row = m0 + wm * 32 + i * 16 + r + rhi;
        int col = n0 + wn * 64 + j * 16 + cl;
        float v = acc[i][j][r];
        if (RELU) v = fmaxf(v, 0.0f);
        size_t idx = (size_t)row * N + col;
        if (Cf) Cf[idx] = v;
        if (Cb) Cb[idx] = (bf16_t)v;
      }
}

// ---------------------------------------------------------------------------
// Attention scores: scores[b,h,q,k] = (Q_h[q,:] . K_h[k,:]) / 8, mask -> -1e9
// Block tile 64(q) x 128(k), contraction over d=64 in 2 steps of 32.
// ---------------------------------------------------------------------------
#define SBM 64
#define SBN 128
#define SLD 34

__global__ __launch_bounds__(256)
void attn_scores_kernel(const bf16_t* __restrict__ Q, const bf16_t* __restrict__ Km,
                        const unsigned char* __restrict__ mask,
                        float* __restrict__ attn) {
  __shared__ bf16_t As[SBM * SLD];
  __shared__ bf16_t Bs[SBN * SLD];

  const int bh = blockIdx.z;              // b*8 + h
  const int b  = bh >> 3;
  const int h  = bh & 7;
  const bf16_t* Qh = Q  + (size_t)b * SEQ * DMODEL + h * DHEAD;
  const bf16_t* Kh = Km + (size_t)b * SEQ * DMODEL + h * DHEAD;

  const int tid  = threadIdx.x;
  const int lane = tid & 31;
  const int wid  = tid >> 5;
  const int wm   = wid >> 2;
  const int wn   = wid & 3;
  const int q0   = blockIdx.y * SBM;
  const int n0   = blockIdx.x * SBN;

  f32x8 acc[2][2] = {};

  for (int k0 = 0; k0 < DHEAD; k0 += 32) {
    // Q tile: 64 x 32 as 4B pairs
#pragma unroll
    for (int i = 0; i < 4; ++i) {
      int p = tid + i * 256;
      int r = p >> 4;
      int c = (p & 15) << 1;
      *(uint32_t*)&As[r * SLD + c] =
          *(const uint32_t*)&Qh[(size_t)(q0 + r) * DMODEL + k0 + c];
    }
    // K tile: Bs[n][d] = Kh[(n0+n)*DMODEL + k0+d]  (contraction dim contiguous)
#pragma unroll
    for (int i = 0; i < 8; ++i) {
      int p = tid + i * 256;
      int n = p >> 4;
      int c = (p & 15) << 1;
      *(uint32_t*)&Bs[n * SLD + c] =
          *(const uint32_t*)&Kh[(size_t)(n0 + n) * DMODEL + k0 + c];
    }
    __syncthreads();

    bf16x16 a0 = load_frag_a(As, SLD, wm * 32 + 0,  lane);
    bf16x16 a1 = load_frag_a(As, SLD, wm * 32 + 16, lane);
    bf16x16 b0 = load_frag_b(Bs, SLD, wn * 32 + 0,  lane);
    bf16x16 b1 = load_frag_b(Bs, SLD, wn * 32 + 16, lane);
    acc[0][0] = wmma_bf16(a0, b0, acc[0][0]);
    acc[0][1] = wmma_bf16(a0, b1, acc[0][1]);
    acc[1][0] = wmma_bf16(a1, b0, acc[1][0]);
    acc[1][1] = wmma_bf16(a1, b1, acc[1][1]);
    __syncthreads();
  }

  const int rhi = (lane >> 4) << 3;
  const int cl  = lane & 15;
  float* out = attn + (size_t)bh * SEQ * SEQ;
#pragma unroll
  for (int i = 0; i < 2; ++i)
#pragma unroll
    for (int j = 0; j < 2; ++j)
#pragma unroll
      for (int r = 0; r < 8; ++r) {
        int q = q0 + wm * 32 + i * 16 + r + rhi;
        int k = n0 + wn * 32 + j * 16 + cl;
        float v = acc[i][j][r] * 0.125f;  // 1/sqrt(64)
        if (mask[((size_t)b * SEQ + q) * SEQ + k]) v = -1e9f;
        out[(size_t)q * SEQ + k] = v;
      }
}

// ---------------------------------------------------------------------------
// Row softmax over last dim (row length SEQ=2048), in place.
// ---------------------------------------------------------------------------
__global__ __launch_bounds__(256)
void softmax_kernel(float* __restrict__ attn) {
  __shared__ float red[256];
  float* p = attn + (size_t)blockIdx.x * SEQ;
  const int tid = threadIdx.x;

  float m = -3.4e38f;
  for (int i = tid; i < SEQ; i += 256) m = fmaxf(m, p[i]);
  red[tid] = m;
  __syncthreads();
  for (int s = 128; s > 0; s >>= 1) {
    if (tid < s) red[tid] = fmaxf(red[tid], red[tid + s]);
    __syncthreads();
  }
  m = red[0];
  __syncthreads();

  float sum = 0.0f;
  for (int i = tid; i < SEQ; i += 256) {
    float e = __expf(p[i] - m);
    p[i] = e;
    sum += e;
  }
  red[tid] = sum;
  __syncthreads();
  for (int s = 128; s > 0; s >>= 1) {
    if (tid < s) red[tid] += red[tid + s];
    __syncthreads();
  }
  float inv = 1.0f / red[0];
  for (int i = tid; i < SEQ; i += 256) p[i] *= inv;
}

// ---------------------------------------------------------------------------
// Context: ctx[b,q,h*64+d] = sum_k attn[b,h,q,k] * V_h[k,d]
// 128 threads = 4 waves (2x2), tile 64(q) x 64(d), BK=32 over keys.
// attn f32 -> bf16 conversion happens while staging into LDS.
// ---------------------------------------------------------------------------
#define CLD 34

__global__ __launch_bounds__(128)
void attn_context_kernel(const float* __restrict__ attn, const bf16_t* __restrict__ V,
                         bf16_t* __restrict__ ctx) {
  __shared__ bf16_t As[64 * CLD];
  __shared__ bf16_t Bs[64 * CLD];

  const int bh = blockIdx.z;
  const int b  = bh >> 3;
  const int h  = bh & 7;
  const float*  Ah = attn + (size_t)bh * SEQ * SEQ;
  const bf16_t* Vh = V + (size_t)b * SEQ * DMODEL + h * DHEAD;

  const int tid  = threadIdx.x;
  const int lane = tid & 31;
  const int wid  = tid >> 5;
  const int wm   = wid >> 1;          // 0..1
  const int wn   = wid & 1;           // 0..1
  const int q0   = blockIdx.y * 64;

  f32x8 acc[2][2] = {};

  for (int k0 = 0; k0 < SEQ; k0 += 32) {
    // attn tile 64q x 32k, f32->bf16 (coalesced over k)
#pragma unroll
    for (int i = 0; i < 16; ++i) {
      int p = tid + i * 128;
      int r = p >> 5;
      int c = p & 31;
      As[r * CLD + c] = (bf16_t)Ah[(size_t)(q0 + r) * SEQ + k0 + c];
    }
    // V tile into [d][k]: Bs[d][k] = Vh[(k0+k)*DMODEL + d] (coalesced over d)
#pragma unroll
    for (int i = 0; i < 16; ++i) {
      int p = tid + i * 128;
      int d = p & 63;
      int k = p >> 6;
      Bs[d * CLD + k] = Vh[(size_t)(k0 + k) * DMODEL + d];
    }
    __syncthreads();

    bf16x16 a0 = load_frag_a(As, CLD, wm * 32 + 0,  lane);
    bf16x16 a1 = load_frag_a(As, CLD, wm * 32 + 16, lane);
    bf16x16 b0 = load_frag_b(Bs, CLD, wn * 32 + 0,  lane);
    bf16x16 b1 = load_frag_b(Bs, CLD, wn * 32 + 16, lane);
    acc[0][0] = wmma_bf16(a0, b0, acc[0][0]);
    acc[0][1] = wmma_bf16(a0, b1, acc[0][1]);
    acc[1][0] = wmma_bf16(a1, b0, acc[1][0]);
    acc[1][1] = wmma_bf16(a1, b1, acc[1][1]);
    __syncthreads();
  }

  const int rhi = (lane >> 4) << 3;
  const int cl  = lane & 15;
#pragma unroll
  for (int i = 0; i < 2; ++i)
#pragma unroll
    for (int j = 0; j < 2; ++j)
#pragma unroll
      for (int r = 0; r < 8; ++r) {
        int q = q0 + wm * 32 + i * 16 + r + rhi;
        int d = wn * 32 + j * 16 + cl;
        ctx[(size_t)(b * SEQ + q) * DMODEL + h * DHEAD + d] = (bf16_t)acc[i][j][r];
      }
}

// ---------------------------------------------------------------------------
// y = LayerNorm(a + b) * g + beta ; row length DMODEL=512, one block per row.
// ---------------------------------------------------------------------------
__global__ __launch_bounds__(256)
void add_ln_kernel(const float* __restrict__ a, const float* __restrict__ b,
                   const float* __restrict__ g, const float* __restrict__ be,
                   float* __restrict__ outf, bf16_t* __restrict__ outb) {
  __shared__ float red[256];
  const size_t row = blockIdx.x;
  const int tid = threadIdx.x;
  const float* pa = a + row * DMODEL;
  const float* pb = b + row * DMODEL;

  float x0 = pa[tid] + pb[tid];
  float x1 = pa[tid + 256] + pb[tid + 256];

  red[tid] = x0 + x1;
  __syncthreads();
  for (int s = 128; s > 0; s >>= 1) {
    if (tid < s) red[tid] += red[tid + s];
    __syncthreads();
  }
  float mu = red[0] * (1.0f / DMODEL);
  __syncthreads();

  red[tid] = x0 * x0 + x1 * x1;
  __syncthreads();
  for (int s = 128; s > 0; s >>= 1) {
    if (tid < s) red[tid] += red[tid + s];
    __syncthreads();
  }
  float var = red[0] * (1.0f / DMODEL) - mu * mu;
  float rs  = rsqrtf(var + 1e-5f);

  float y0 = (x0 - mu) * rs * g[tid]       + be[tid];
  float y1 = (x1 - mu) * rs * g[tid + 256] + be[tid + 256];
  outf[row * DMODEL + tid]       = y0;
  outf[row * DMODEL + tid + 256] = y1;
  if (outb) {
    outb[row * DMODEL + tid]       = (bf16_t)y0;
    outb[row * DMODEL + tid + 256] = (bf16_t)y1;
  }
}

// ---------------------------------------------------------------------------
// Host launcher
// ---------------------------------------------------------------------------
extern "C" void kernel_launch(void* const* d_in, const int* in_sizes, int n_in,
                              void* d_out, int out_size, void* d_ws, size_t ws_size,
                              hipStream_t stream) {
  const float*         x    = (const float*)d_in[0];
  const unsigned char* mask = (const unsigned char*)d_in[1];
  const float* Wq   = (const float*)d_in[2];
  const float* Wk   = (const float*)d_in[3];
  const float* Wv   = (const float*)d_in[4];
  const float* Wo   = (const float*)d_in[5];
  const float* ln1g = (const float*)d_in[6];
  const float* ln1b = (const float*)d_in[7];
  const float* W1   = (const float*)d_in[8];
  const float* W2   = (const float*)d_in[9];
  const float* ln2g = (const float*)d_in[10];
  const float* ln2b = (const float*)d_in[11];

  float* out  = (float*)d_out;                       // [4,2048,512]
  float* attn = out + (size_t)MTOT * DMODEL;         // [4,8,2048,2048]

  // --- workspace carve-up (~90 MB total, regions reused) ---
  char* ws = (char*)d_ws;
  size_t o = 0;
  bf16_t* Xbf   = (bf16_t*)(ws + o); o += (size_t)MTOT * DMODEL * 2;
  bf16_t* Wqbf  = (bf16_t*)(ws + o); o += (size_t)DMODEL * DMODEL * 2;
  bf16_t* Wkbf  = (bf16_t*)(ws + o); o += (size_t)DMODEL * DMODEL * 2;
  bf16_t* Wvbf  = (bf16_t*)(ws + o); o += (size_t)DMODEL * DMODEL * 2;
  bf16_t* Wobf  = (bf16_t*)(ws + o); o += (size_t)DMODEL * DMODEL * 2;
  bf16_t* W1bf  = (bf16_t*)(ws + o); o += (size_t)DMODEL * DFF * 2;
  bf16_t* W2bf  = (bf16_t*)(ws + o); o += (size_t)DFF * DMODEL * 2;
  bf16_t* Qbf   = (bf16_t*)(ws + o); o += (size_t)MTOT * DMODEL * 2;
  bf16_t* Kbf   = (bf16_t*)(ws + o); o += (size_t)MTOT * DMODEL * 2;
  bf16_t* Vbf   = (bf16_t*)(ws + o); o += (size_t)MTOT * DMODEL * 2;
  bf16_t* ctxbf = (bf16_t*)(ws + o); o += (size_t)MTOT * DMODEL * 2;
  float*  projf = (float*)(ws + o);  o += (size_t)MTOT * DMODEL * 4;
  float*  aoutf = (float*)(ws + o);  o += (size_t)MTOT * DMODEL * 4;
  bf16_t* aoutb = (bf16_t*)(ws + o); o += (size_t)MTOT * DMODEL * 2;
  bf16_t* ff1bf = Qbf;    // reuse Q/K/V/ctx (32 MB) after attention is done
  float*  ff2f  = projf;  // reuse proj after LN1

  // --- 1) f32 -> bf16 conversions ---
  auto cvt = [&](const float* s, bf16_t* d, int n) {
    cvt_f32_bf16_kernel<<<(n + 255) / 256, 256, 0, stream>>>(s, d, n);
  };
  cvt(x,  Xbf,  MTOT * DMODEL);
  cvt(Wq, Wqbf, DMODEL * DMODEL);
  cvt(Wk, Wkbf, DMODEL * DMODEL);
  cvt(Wv, Wvbf, DMODEL * DMODEL);
  cvt(Wo, Wobf, DMODEL * DMODEL);
  cvt(W1, W1bf, DMODEL * DFF);
  cvt(W2, W2bf, DFF * DMODEL);

  // --- 2) Q, K, V projections: [8192,512] x [512,512] -> bf16 ---
  dim3 gQKV(DMODEL / GBN, MTOT / GBM);   // (4, 64)
  gemm_bf16_kernel<false><<<gQKV, 256, 0, stream>>>(Xbf, Wqbf, nullptr, Qbf,
                                                    MTOT, DMODEL, DMODEL);
  gemm_bf16_kernel<false><<<gQKV, 256, 0, stream>>>(Xbf, Wkbf, nullptr, Kbf,
                                                    MTOT, DMODEL, DMODEL);
  gemm_bf16_kernel<false><<<gQKV, 256, 0, stream>>>(Xbf, Wvbf, nullptr, Vbf,
                                                    MTOT, DMODEL, DMODEL);

  // --- 3) scores (+scale +mask) into attn region of d_out ---
  dim3 gS(SEQ / SBN, SEQ / SBM, BATCH * NHEADS);  // (16, 32, 32)
  attn_scores_kernel<<<gS, 256, 0, stream>>>(Qbf, Kbf, mask, attn);

  // --- 4) softmax in place ---
  softmax_kernel<<<BATCH * NHEADS * SEQ, 256, 0, stream>>>(attn);

  // --- 5) context = attn @ V -> ctxbf [8192,512] ---
  dim3 gC(1, SEQ / 64, BATCH * NHEADS);           // (1, 32, 32)
  attn_context_kernel<<<gC, 128, 0, stream>>>(attn, Vbf, ctxbf);

  // --- 6) output projection + residual + LN1 ---
  gemm_bf16_kernel<false><<<gQKV, 256, 0, stream>>>(ctxbf, Wobf, projf, nullptr,
                                                    MTOT, DMODEL, DMODEL);
  add_ln_kernel<<<MTOT, 256, 0, stream>>>(projf, x, ln1g, ln1b, aoutf, aoutb);

  // --- 7) FFN: relu(aout @ W1) @ W2 ---
  dim3 gF1(DFF / GBN, MTOT / GBM);                // (16, 64)
  gemm_bf16_kernel<true><<<gF1, 256, 0, stream>>>(aoutb, W1bf, nullptr, ff1bf,
                                                  MTOT, DFF, DMODEL);
  dim3 gF2(DMODEL / GBN, MTOT / GBM);             // (4, 64)
  gemm_bf16_kernel<false><<<gF2, 256, 0, stream>>>(ff1bf, W2bf, ff2f, nullptr,
                                                   MTOT, DMODEL, DFF);

  // --- 8) residual + LN2 -> final output ---
  add_ln_kernel<<<MTOT, 256, 0, stream>>>(ff2f, aoutf, ln2g, ln2b, out, nullptr);
}
